// Dense_4277787427179
// MI455X (gfx1250) — matrix operand order repaired and checked by reading, not compiled, not run on hardware
//
#include <hip/hip_runtime.h>

// ---------------------------------------------------------------------------
// MI455X (gfx1250) split-bf16 WMMA implementation of the per-degree dense mix.
//   out[b,p, c in seg_l, :] = x[b,p,seg_l,:] @ W_{e|o}[l]  (+ bias for l=0,p=0)
// HBM-bound target: ~1.07 GiB traffic @ 23.3 TB/s. f32 precision recovered via
// hi/lo bf16 decomposition -> 3x v_wmma_f32_16x16x32_bf16 per tile.
// A-tile converted to bf16 hi/lo ONCE during LDS staging. GEMM body templated
// on degree L so all /d address math lowers to magic-multiply (no v_rcp divs).
// ---------------------------------------------------------------------------

typedef __attribute__((ext_vector_type(16))) __bf16 v16bf;
typedef __attribute__((ext_vector_type(8)))  __bf16 v8bf;
typedef __attribute__((ext_vector_type(8)))  float  v8f;
typedef __attribute__((ext_vector_type(4)))  float  v4f;

#define LDSTRIDE 264             // bf16 units: 256 + 8 pad (528B rows, bank rotate)
#define NWELEM   (8 * 256 * 256) // elements per weight plane (p,l,n,k)

union BFrag { v16bf v; v4f q[2]; };   // 32B: two 16B chunks

// ---------------------------------------------------------------------------
// Prologue: split W_e/W_o (f32, [l][k][n]) into bf16 hi/lo planes, transposed
// to [p][l][n][k] so B fragments are K-contiguous 16-byte loads.
// ---------------------------------------------------------------------------
__global__ __launch_bounds__(256) void wsplit_kernel(
    const float* __restrict__ We, const float* __restrict__ Wo,
    __bf16* __restrict__ Whi, __bf16* __restrict__ Wlo) {
    int blk = blockIdx.x;              // (p*4 + l)*256 + k
    int n   = threadIdx.x;
    int k   = blk & 255;
    int pl  = blk >> 8;                // p*4 + l
    const float* W = ((pl >> 2) == 0 ? We : Wo) + (pl & 3) * 65536;
    float w = W[k * 256 + n];
    __bf16 h = (__bf16)w;
    int idx = (pl * 256 + n) * 256 + k;
    Whi[idx] = h;
    Wlo[idx] = (__bf16)(w - (float)h);
}

// ---------------------------------------------------------------------------
// GEMM body, specialized per degree L (d = 2L+1 becomes compile-time).
// ---------------------------------------------------------------------------
template <int L>
__device__ __forceinline__ void seg_gemm_body(
    const float* __restrict__ x, const float* __restrict__ bias,
    const __bf16* __restrict__ Whi, const __bf16* __restrict__ Wlo,
    float* __restrict__ out, int p, int m0,
    __bf16* Ahi_s, __bf16* Alo_s) {

    constexpr int D = 2 * L + 1;
    // Row i of this (p,L) GEMM -> flat float offset in (B,2,16,256) layout.
    auto row_off = [&](int i) -> int {
        int b = (unsigned)i / (unsigned)D;        // constant divisor -> mul/shift
        int c = L * L + (i - b * D);
        return ((b * 2 + p) * 16 + c) << 8;       // * 256 floats
    };

    // ---- stage 64x256 f32 A-tile as bf16 hi/lo into LDS (split once) ----
    {
        int tid = threadIdx.x;
        int r   = tid >> 2;            // 64 rows, 4 threads per row
        int cc  = (tid & 3) * 64;      // 64-element column chunk
        const v4f* src = (const v4f*)(x + row_off(m0 + r) + cc);
        __bf16* dh = Ahi_s + r * LDSTRIDE + cc;
        __bf16* dl = Alo_s + r * LDSTRIDE + cc;
        #pragma unroll
        for (int j = 0; j < 8; ++j) {
            v4f a = __builtin_nontemporal_load(src + 2 * j);
            v4f b = __builtin_nontemporal_load(src + 2 * j + 1);
            float fv[8] = {a[0], a[1], a[2], a[3], b[0], b[1], b[2], b[3]};
            v8bf h, lo;
            #pragma unroll
            for (int e = 0; e < 8; ++e) {
                __bf16 hh = (__bf16)fv[e];
                h[e]  = hh;
                lo[e] = (__bf16)(fv[e] - (float)hh);
            }
            *(v8bf*)(dh + j * 8) = h;
            *(v8bf*)(dl + j * 8) = lo;
        }
    }
    __syncthreads();

    int wave  = threadIdx.x >> 5;      // 8 waves, each owns 32 output columns
    int lane  = threadIdx.x & 31;
    int lrow  = lane & 15;
    int khalf = lane >> 4;             // lane half selects K sub-range
    int nbase = wave * 32;

    const __bf16* WHbase = Whi + (p * 4 + L) * 65536;
    const __bf16* WLbase = Wlo + (p * 4 + L) * 65536;

    v8f acc[4][2] = {};                // 4 M-subtiles x 2 N-subtiles

    for (int k0 = 0; k0 < 256; k0 += 32) {
        // ---- B fragments (32x16, K-contiguous per lane: K = khalf*16 + e) ----
        BFrag bh[2], bl[2];
        #pragma unroll
        for (int nt = 0; nt < 2; ++nt) {
            int n = nbase + nt * 16 + lrow;
            const __bf16* ph = WHbase + n * 256 + k0 + khalf * 16;
            const __bf16* pq = WLbase + n * 256 + k0 + khalf * 16;
            bh[nt].q[0] = *(const v4f*)(ph);
            bh[nt].q[1] = *(const v4f*)(ph + 8);
            bl[nt].q[0] = *(const v4f*)(pq);
            bl[nt].q[1] = *(const v4f*)(pq + 8);
        }
        #pragma unroll
        for (int m = 0; m < 4; ++m) {
            // ---- A fragments (16x32): lane half -> K {0-15} / {16-31}
            int aoff = (m * 16 + lrow) * LDSTRIDE + k0 + khalf * 16;
            BFrag ah, al;
            ah.q[0] = *(const v4f*)(Ahi_s + aoff);        // K +0..7   (8 bf16)
            ah.q[1] = *(const v4f*)(Ahi_s + aoff + 8);    // K +8..15
            al.q[0] = *(const v4f*)(Alo_s + aoff);
            al.q[1] = *(const v4f*)(Alo_s + aoff + 8);
            // ---- bf16x3: Alo*Bhi + Ahi*Blo + Ahi*Bhi, f32 accumulate ----
            #pragma unroll
            for (int nt = 0; nt < 2; ++nt) {
                acc[m][nt] = __builtin_amdgcn_wmma_f32_16x16x32_bf16(
                    false, al.v, false, bh[nt].v, (short)0, acc[m][nt], false, false);
                acc[m][nt] = __builtin_amdgcn_wmma_f32_16x16x32_bf16(
                    false, ah.v, false, bl[nt].v, (short)0, acc[m][nt], false, false);
                acc[m][nt] = __builtin_amdgcn_wmma_f32_16x16x32_bf16(
                    false, ah.v, false, bh[nt].v, (short)0, acc[m][nt], false, false);
            }
        }
    }

    // ---- epilogue: bias (only l=0, parity even) + NT stores in x layout ----
    float bs[2] = {0.f, 0.f};
    if (L == 0 && p == 0) {
        bs[0] = bias[nbase + lrow];
        bs[1] = bias[nbase + 16 + lrow];
    }
    #pragma unroll
    for (int m = 0; m < 4; ++m) {
        #pragma unroll
        for (int j = 0; j < 8; ++j) {
            int row = m * 16 + khalf * 8 + j;   // C/D layout: VGPR j -> M=j / j+8
            int ro  = row_off(m0 + row);
            #pragma unroll
            for (int nt = 0; nt < 2; ++nt) {
                int col = nbase + nt * 16 + lrow;
                __builtin_nontemporal_store(acc[m][nt][j] + bs[nt], out + ro + col);
            }
        }
    }
}

// ---------------------------------------------------------------------------
// Main kernel: one 8-wave block per (p, l, 64-row tile); uniform dispatch on l.
// ---------------------------------------------------------------------------
__global__ __launch_bounds__(256) void seg_gemm_kernel(
    const float* __restrict__ x, const float* __restrict__ bias,
    const __bf16* __restrict__ Whi, const __bf16* __restrict__ Wlo,
    float* __restrict__ out) {

    __shared__ __align__(16) __bf16 Ahi_s[64 * LDSTRIDE];   // 33 KB
    __shared__ __align__(16) __bf16 Alo_s[64 * LDSTRIDE];   // 33 KB

    int bid = blockIdx.x;              // 0..8191
    int p = bid >> 12;                 // parity
    int t = bid & 4095;                // tile id within parity

    if (t < 256) {
        seg_gemm_body<0>(x, bias, Whi, Wlo, out, p, t * 64, Ahi_s, Alo_s);
    } else if (t < 1024) {
        seg_gemm_body<1>(x, bias, Whi, Wlo, out, p, (t - 256) * 64, Ahi_s, Alo_s);
    } else if (t < 2304) {
        seg_gemm_body<2>(x, bias, Whi, Wlo, out, p, (t - 1024) * 64, Ahi_s, Alo_s);
    } else {
        seg_gemm_body<3>(x, bias, Whi, Wlo, out, p, (t - 2304) * 64, Ahi_s, Alo_s);
    }
}

// ---------------------------------------------------------------------------
extern "C" void kernel_launch(void* const* d_in, const int* in_sizes, int n_in,
                              void* d_out, int out_size, void* d_ws, size_t ws_size,
                              hipStream_t stream) {
    const float* x    = (const float*)d_in[0];
    const float* We   = (const float*)d_in[1];
    const float* Wo   = (const float*)d_in[2];
    const float* bias = (const float*)d_in[3];
    float* out = (float*)d_out;

    __bf16* Whi = (__bf16*)d_ws;           // 1 MiB
    __bf16* Wlo = Whi + NWELEM;            // 1 MiB

    // Split/transpose weights into workspace (L2-resident thereafter).
    wsplit_kernel<<<2048, 256, 0, stream>>>(We, Wo, Whi, Wlo);

    // 8192 blocks: 2 parities x (256 + 768 + 1280 + 1792) 64-row tiles.
    seg_gemm_kernel<<<8192, 256, 0, stream>>>(x, bias, Whi, Wlo, out);
}
